// CachedMPS_82824149336270
// MI455X (gfx1250) — matrix operand order repaired and checked by reading, not compiled, not run on hardware
//
#include <hip/hip_runtime.h>

typedef _Float16 v16h __attribute__((ext_vector_type(16)));
typedef _Float16 h2   __attribute__((ext_vector_type(2)));
typedef __fp16   f16x2 __attribute__((ext_vector_type(2)));   // cvt_pkrtz result type
typedef float    v8f  __attribute__((ext_vector_type(8)));

#define HALF_PI 1.5707963267948966f

constexpr int B    = 32768;
constexpr int L    = 256;
constexpr int D    = 32;
constexpr int C    = 10;
constexpr int NMID = L - 2;           // 254 middle sites
constexpr int NOPS = NMID * 4 + 2;    // 4 A-operand chunks per step + 2 readout chunks
// One operand chunk = 32 lanes x 16 f16 (per-lane WMMA A-operand, contiguous k order) = 1 KB.

// Half-wave swap (lane <-> lane^16) as a pure-VALU v_permlanex16_b32 with
// identity row selects — replaces ds_bpermute-based __shfl_xor(…,16).
__device__ __forceinline__ int xswap16_i(int v) {
  return __builtin_amdgcn_permlanex16(v, v, 0x76543210, 0xfedcba98, true, false);
}
__device__ __forceinline__ float xswap16_f(float v) {
  int i = __builtin_bit_cast(int, v);
  return __builtin_bit_cast(float, xswap16_i(i));
}

// ---------------------------------------------------------------------------
// Pre-swizzle weights into per-lane WMMA A-operand layout (f32 -> f16).
// A order used here (contiguous k, same as B): lane L holds row r = L%16; element j holds
//   k = k0 + j,  k0 = (L>=16)*16.
// Operand q at step l: q&1 = feature f, q>>1 = output col tile t.
// Value = A_l[f][k][c] with c = r + 16*t  (A^T restricted to rows of tile t).
// ---------------------------------------------------------------------------
__global__ void prep_weights(const float* __restrict__ core_mid,
                             const float* __restrict__ coreN,
                             _Float16* __restrict__ w) {
  int t = blockIdx.x * blockDim.x + threadIdx.x;
  int opIdx = t >> 5, lane = t & 31;
  if (opIdx >= NOPS) return;
  int r  = lane & 15;
  int k0 = (lane >> 4) * 16;
  v16h out;
  if (opIdx < NMID * 4) {
    int l = opIdx >> 2, q = opIdx & 3;
    int f = q & 1, tile = q >> 1;
    int c = r + 16 * tile;
#pragma unroll
    for (int j = 0; j < 16; ++j) {
      int k = k0 + j;
      out[j] = (_Float16)core_mid[(((size_t)l * 2 + f) * D + k) * D + c];
    }
  } else {                               // readout operands: rows 10..15 zero
    int f = opIdx - NMID * 4;
#pragma unroll
    for (int j = 0; j < 16; ++j) {
      int k = k0 + j;
      out[j] = (r < C) ? (_Float16)coreN[((size_t)f * D + k) * C + r]
                       : (_Float16)0.0f;
    }
  }
  *(v16h*)(w + (size_t)opIdx * 512 + lane * 16) = out;   // two b128 stores
}

// ---------------------------------------------------------------------------
// Transpose x[B,L] -> feats[L][B] as packed (cos, sin) f16 pairs.
// ---------------------------------------------------------------------------
__global__ void prep_feats(const float* __restrict__ x, h2* __restrict__ feats) {
  size_t t = (size_t)blockIdx.x * blockDim.x + threadIdx.x;   // t = l*B + b
  if (t >= (size_t)B * L) return;
  int l = (int)(t / B), b = (int)(t % B);
  float xv = x[(size_t)b * L + l];
  h2 f;
  f.x = (_Float16)__cosf(HALF_PI * xv);
  f.y = (_Float16)__sinf(HALF_PI * xv);
  feats[t] = f;
}

// ---------------------------------------------------------------------------
// Main MPS scan. One wave handles 16 batch rows (N axis of WMMA).
// M kept as f16 B-operand (k = bond, n = batch); weights stream as A-operands.
// ---------------------------------------------------------------------------
template <bool USE_FEATS>
__global__ __launch_bounds__(256) void mps_kernel(
    const float* __restrict__ x, const float* __restrict__ core0,
    const h2* __restrict__ feats, const _Float16* __restrict__ w,
    float* __restrict__ out) {
  const int  lane = threadIdx.x & 31;
  const int  wid  = threadIdx.x >> 5;
  const int  n    = lane & 15;
  const bool hi   = lane >= 16;
  const int  b    = (blockIdx.x * 8 + wid) * 16 + n;

  const v16h* wv = (const v16h*)w;     // [opIdx][lane] chunks of 16 f16
  const v8f   zero = {};

  // ---- site 0: M0[b,a] = c0*core0[0,0,a] + s0*core0[1,0,a], B-layout a=e+16*hi
  float c0, s0;
  if (USE_FEATS) {
    h2 f0 = feats[b];
    c0 = (float)f0.x; s0 = (float)f0.y;
  } else {
    float xv = x[(size_t)b * L];
    c0 = __cosf(HALF_PI * xv); s0 = __sinf(HALF_PI * xv);
  }
  v16h mb;
#pragma unroll
  for (int e = 0; e < 16; ++e) {
    int a = e + (hi ? 16 : 0);
    mb[e] = (_Float16)(c0 * core0[a] + s0 * core0[D + a]);
  }

  // current-step weight operands (double-buffered against next step)
  v16h Wc0 = wv[(size_t)0 * 32 + lane], Wc1 = wv[(size_t)1 * 32 + lane];
  v16h Wc2 = wv[(size_t)2 * 32 + lane], Wc3 = wv[(size_t)3 * 32 + lane];

  for (int l = 0; l < NMID; ++l) {
    // prefetch next step's operands (last iter: readout operands)
    size_t nb = (l + 1 < NMID) ? (size_t)(l + 1) * 4 : (size_t)NMID * 4;
    v16h Wn0 = wv[(nb + 0) * 32 + lane];
    v16h Wn1 = wv[(nb + 1) * 32 + lane];
    v16h Wn2 = Wn0, Wn3 = Wn1;
    if (l + 1 < NMID) {
      Wn2 = wv[(nb + 2) * 32 + lane];
      Wn3 = wv[(nb + 3) * 32 + lane];
    }

    // per-batch feature scalars for this site
    _Float16 fc, fs;
    if (USE_FEATS) {
      h2 f = feats[(size_t)(l + 1) * B + b];
      fc = f.x; fs = f.y;
    } else {
      float xv = x[(size_t)b * L + (l + 1)];
      fc = (_Float16)__cosf(HALF_PI * xv);
      fs = (_Float16)__sinf(HALF_PI * xv);
    }
    v16h bv0 = mb * fc;        // v_pk_mul_f16 x8
    v16h bv1 = mb * fs;

    // M_new^T = A0^T (v0.M^T) + A1^T (v1.M^T), two 16-row output tiles
    v8f acc0 = __builtin_amdgcn_wmma_f32_16x16x32_f16(false, Wc0, false, bv0,
                                                      (short)0, zero, false, false);
    acc0     = __builtin_amdgcn_wmma_f32_16x16x32_f16(false, Wc1, false, bv1,
                                                      (short)0, acc0, false, false);
    v8f acc1 = __builtin_amdgcn_wmma_f32_16x16x32_f16(false, Wc2, false, bv0,
                                                      (short)0, zero, false, false);
    acc1     = __builtin_amdgcn_wmma_f32_16x16x32_f16(false, Wc3, false, bv1,
                                                      (short)0, acc1, false, false);

    // row (= batch column) L2 norm: column n split across lanes n and n+16.
    // Explicit log-depth tree keeps the dependence chain short.
    float p0 = acc0[0] * acc0[0] + acc1[0] * acc1[0];
    float p1 = acc0[1] * acc0[1] + acc1[1] * acc1[1];
    float p2 = acc0[2] * acc0[2] + acc1[2] * acc1[2];
    float p3 = acc0[3] * acc0[3] + acc1[3] * acc1[3];
    float p4 = acc0[4] * acc0[4] + acc1[4] * acc1[4];
    float p5 = acc0[5] * acc0[5] + acc1[5] * acc1[5];
    float p6 = acc0[6] * acc0[6] + acc1[6] * acc1[6];
    float p7 = acc0[7] * acc0[7] + acc1[7] * acc1[7];
    float ss = ((p0 + p1) + (p2 + p3)) + ((p4 + p5) + (p6 + p7));
    ss += xswap16_f(ss);
    // raw v_sqrt_f32 / v_rcp_f32 (no range-fixup code); plenty for f16 data
    float rinv = __builtin_amdgcn_rcpf(__builtin_amdgcn_sqrtf(ss) + 1e-8f);

    // D-layout -> next B-layout. Convert to packed f16 pairs FIRST, then do
    // the half-wave swap on 32-bit packed values: 8 cvt_pk + 8 permlanex + 8 sel.
    // Adjacent target k (e=2j,2j+1) always come from the same source lane and
    // same accumulator, so packing before the swap is layout-exact.
#pragma unroll
    for (int j = 0; j < 4; ++j) {
      f16x2 pka = __builtin_amdgcn_cvt_pkrtz(acc0[2 * j] * rinv, acc0[2 * j + 1] * rinv);
      f16x2 pkb = __builtin_amdgcn_cvt_pkrtz(acc1[2 * j] * rinv, acc1[2 * j + 1] * rinv);
      int ia = __builtin_bit_cast(int, pka);
      int ib = __builtin_bit_cast(int, pkb);
      int xa = xswap16_i(ia);               // executed by all lanes
      int xb = xswap16_i(ib);
      h2 lp = __builtin_bit_cast(h2, hi ? xb : ia);  // k = 2j,2j+1   (+16*hi)
      h2 hp = __builtin_bit_cast(h2, hi ? ib : xa);  // k = 8+2j,9+2j (+16*hi)
      mb[2 * j]     = lp.x;
      mb[2 * j + 1] = lp.y;
      mb[8 + 2 * j] = hp.x;
      mb[9 + 2 * j] = hp.y;
    }

    Wc0 = Wn0; Wc1 = Wn1; Wc2 = Wn2; Wc3 = Wn3;
  }

  // ---- readout: logits^T = coreN0^T (v0.M^T) + coreN1^T (v1.M^T), C=10 rows
  _Float16 fc, fs;
  if (USE_FEATS) {
    h2 f = feats[(size_t)(L - 1) * B + b];
    fc = f.x; fs = f.y;
  } else {
    float xv = x[(size_t)b * L + (L - 1)];
    fc = (_Float16)__cosf(HALF_PI * xv);
    fs = (_Float16)__sinf(HALF_PI * xv);
  }
  v16h bv0 = mb * fc;
  v16h bv1 = mb * fs;
  v8f df = __builtin_amdgcn_wmma_f32_16x16x32_f16(false, Wc0, false, bv0,
                                                  (short)0, zero, false, false);
  df     = __builtin_amdgcn_wmma_f32_16x16x32_f16(false, Wc1, false, bv1,
                                                  (short)0, df, false, false);

  // D element (c, n): lane = n + 16*(c>=8), vgpr = c%8 ; keep c < 10
  if (!hi) {
#pragma unroll
    for (int u = 0; u < 8; ++u) out[(size_t)b * C + u] = df[u];
  } else {
    out[(size_t)b * C + 8] = df[0];
    out[(size_t)b * C + 9] = df[1];
  }
}

// ---------------------------------------------------------------------------
extern "C" void kernel_launch(void* const* d_in, const int* in_sizes, int n_in,
                              void* d_out, int out_size, void* d_ws,
                              size_t ws_size, hipStream_t stream) {
  (void)in_sizes; (void)n_in; (void)out_size;
  const float* x         = (const float*)d_in[0];
  const float* core0     = (const float*)d_in[1];
  const float* cores_mid = (const float*)d_in[2];
  const float* coreN     = (const float*)d_in[3];
  float* out = (float*)d_out;

  _Float16* w = (_Float16*)d_ws;
  size_t wbytes     = (size_t)NOPS * 512 * sizeof(_Float16);    // ~1.02 MB
  size_t featsOff   = (wbytes + 255) & ~(size_t)255;
  size_t featsBytes = (size_t)B * L * sizeof(h2);               // 32 MB
  bool use_feats = ws_size >= featsOff + featsBytes;
  h2* feats = (h2*)((char*)d_ws + featsOff);

  {
    int nthr = NOPS * 32;
    prep_weights<<<(nthr + 255) / 256, 256, 0, stream>>>(cores_mid, coreN, w);
  }
  if (use_feats) {
    size_t tot = (size_t)B * L;
    prep_feats<<<(unsigned)((tot + 255) / 256), 256, 0, stream>>>(x, feats);
    mps_kernel<true><<<B / 128, 256, 0, stream>>>(x, core0, feats, w, out);
  } else {
    mps_kernel<false><<<B / 128, 256, 0, stream>>>(x, core0, nullptr, w, out);
  }
}
